// ScaledDotProductAttention_25340307046955
// MI455X (gfx1250) — matrix operand order repaired and checked by reading, not compiled, not run on hardware
//
#include <hip/hip_runtime.h>

typedef __attribute__((ext_vector_type(16))) _Float16 v16h;
typedef __attribute__((ext_vector_type(8)))  float    v8f;

#define WMMA_F16(A, Bm, C) \
  __builtin_amdgcn_wmma_f32_16x16x32_f16(false, (A), false, (Bm), (short)0, (C), false, false)

constexpr int   BATCH = 32;
constexpr int   SEQ   = 2048;
constexpr int   DIM   = 64;
constexpr int   WAVES = 8;               // waves per block (wave32)
constexpr float SCALE = 0.125f;          // 1/sqrt(64)
constexpr float BIAS  = 10000.0f;        // MASK_BIAS from reference

__global__ __launch_bounds__(WAVES * 32)
void fa_causal_wmma_kernel(const float* __restrict__ Q,
                           const float* __restrict__ K,
                           const float* __restrict__ V,
                           const int*   __restrict__ mask,
                           float*       __restrict__ out)
{
    __shared__ _Float16 ldsP[WAVES * 16 * 32];   // per-wave 16x32 f16 P tile

    const int lane = threadIdx.x & 31;
    const int wave = threadIdx.x >> 5;
    const int qt   = blockIdx.x * WAVES + wave;   // global 16-row q tile id
    const int b    = qt / (SEQ / 16);
    const int q0   = (qt % (SEQ / 16)) * 16;

    const int n    = lane & 15;   // column id inside 16-wide tiles
    const int h    = lane >> 4;   // which half of the wave
    const int roff = h * 8;       // C-fragment row offset for this lane

    // ---------------- Q A-fragments (pre-scaled, f32 -> f16) ----------------
    // A 16x32 in this code's contiguous k order (B matches): lane<16 holds K {0..15}, lane>=16 holds {16..31}
    v16h aQ[2];
    {
        const float* qp = Q + ((size_t)b * SEQ + q0 + n) * DIM;
        #pragma unroll
        for (int f = 0; f < 2; ++f) {
            const int kb = f * 32 + h * 16;
            #pragma unroll
            for (int j = 0; j < 8; ++j) {
                aQ[f][j]     = (_Float16)(qp[kb + j]      * SCALE);
                aQ[f][j + 8] = (_Float16)(qp[kb + 8 + j] * SCALE);
            }
        }
    }

    // ---------------- flash-attention state ----------------
    v8f o[4];
    #pragma unroll
    for (int c = 0; c < 4; ++c)
        o[c] = (v8f){0.f, 0.f, 0.f, 0.f, 0.f, 0.f, 0.f, 0.f};
    float rm[8], rs[8];
    #pragma unroll
    for (int r = 0; r < 8; ++r) { rm[r] = -1e30f; rs[r] = 0.f; }

    _Float16* myP = ldsP + wave * (16 * 32);
    const int kend = q0 + 16;                      // causal horizon (exclusive)

    for (int kt = 0; kt < kend; kt += 32) {
        // ---- K B-fragments: B[kk][n] = K[kt+sub*16+n][kb+kk] ----
        // per lane: 16 consecutive floats of one K row
        v16h bK[2][2];
        #pragma unroll
        for (int sub = 0; sub < 2; ++sub) {
            const float* kp = K + ((size_t)b * SEQ + kt + sub * 16 + n) * DIM + h * 16;
            #pragma unroll
            for (int f = 0; f < 2; ++f)
                #pragma unroll
                for (int j = 0; j < 16; ++j)
                    bK[sub][f][j] = (_Float16)kp[f * 32 + j];
        }
        if (kt + 32 < kend) {
            __builtin_prefetch(K + ((size_t)b * SEQ + kt + 32 + n) * DIM, 0, 0);
            __builtin_prefetch(V + ((size_t)b * SEQ + kt + 32 + n) * DIM, 0, 0);
        }

        // ---- S = (Q*scale) @ K^T : two 16x16 fp32 tiles ----
        v8f s0 = (v8f){0.f,0.f,0.f,0.f,0.f,0.f,0.f,0.f};
        v8f s1 = (v8f){0.f,0.f,0.f,0.f,0.f,0.f,0.f,0.f};
        s0 = WMMA_F16(aQ[0], bK[0][0], s0);
        s0 = WMMA_F16(aQ[1], bK[0][1], s0);
        s1 = WMMA_F16(aQ[0], bK[1][0], s1);
        s1 = WMMA_F16(aQ[1], bK[1][1], s1);

        // ---- causal + padding bias (matches reference: subtract 10000 each) ----
        const bool mv0 = mask[(size_t)b * SEQ + kt + n] != 0;
        const bool mv1 = mask[(size_t)b * SEQ + kt + 16 + n] != 0;
        float tmax[8];
        #pragma unroll
        for (int r = 0; r < 8; ++r) {
            const int rowg = q0 + r + roff;
            float e0 = s0[r], e1 = s1[r];
            if (kt + n      > rowg) e0 -= BIAS;
            if (kt + 16 + n > rowg) e1 -= BIAS;
            if (!mv0) e0 -= BIAS;
            if (!mv1) e1 -= BIAS;
            s0[r] = e0; s1[r] = e1;
            tmax[r] = fmaxf(e0, e1);
        }

        // ---- row max across the 16 lanes holding one row ----
        #pragma unroll
        for (int r = 0; r < 8; ++r) {
            #pragma unroll
            for (int off = 1; off < 16; off <<= 1)
                tmax[r] = fmaxf(tmax[r], __shfl_xor(tmax[r], off, 32));
        }

        float alpha[8], tsum[8];
        #pragma unroll
        for (int r = 0; r < 8; ++r) {
            const float nm = fmaxf(rm[r], tmax[r]);
            alpha[r] = __expf(rm[r] - nm);
            rm[r]    = nm;
            const float p0 = __expf(s0[r] - nm);
            const float p1 = __expf(s1[r] - nm);
            s0[r] = p0; s1[r] = p1;
            tsum[r] = p0 + p1;
        }
        #pragma unroll
        for (int r = 0; r < 8; ++r) {
            #pragma unroll
            for (int off = 1; off < 16; off <<= 1)
                tsum[r] += __shfl_xor(tsum[r], off, 32);
            rs[r] = rs[r] * alpha[r] + tsum[r];
        }
        #pragma unroll
        for (int c = 0; c < 4; ++c)
            #pragma unroll
            for (int r = 0; r < 8; ++r)
                o[c][r] *= alpha[r];

        // ---- P tile (16x32 f16) to LDS: C-layout -> row-major ----
        #pragma unroll
        for (int r = 0; r < 8; ++r) {
            const int row = r + roff;
            myP[row * 32 + n]      = (_Float16)s0[r];
            myP[row * 32 + 16 + n] = (_Float16)s1[r];
        }
        asm volatile("s_wait_dscnt 0" ::: "memory");

        // ---- read P back as A-fragment (row-major, contiguous b128 chunks) ----
        v16h aP;
        {
            const _Float16* pr = myP + (lane & 15) * 32 + h * 16;
            #pragma unroll
            for (int j = 0; j < 8; ++j) {
                aP[j]     = pr[j];
                aP[j + 8] = pr[8 + j];
            }
        }

        // ---- V B-fragments + P@V accumulation ----
        #pragma unroll
        for (int c = 0; c < 4; ++c) {
            v16h bV;
            const float* vp = V + ((size_t)b * SEQ + kt + h * 16) * DIM + c * 16 + n;
            #pragma unroll
            for (int i = 0; i < 8; ++i) {
                bV[2 * i]     = (_Float16)vp[(2 * i) * DIM];
                bV[2 * i + 1] = (_Float16)vp[(2 * i + 1) * DIM];
            }
            o[c] = WMMA_F16(aP, bV, o[c]);
        }
    }

    // ---------------- normalize and write out ----------------
    #pragma unroll
    for (int r = 0; r < 8; ++r) {
        const float inv = 1.0f / rs[r];
        const size_t rowbase = ((size_t)b * SEQ + q0 + r + roff) * DIM;
        #pragma unroll
        for (int c = 0; c < 4; ++c)
            out[rowbase + c * 16 + n] = o[c][r] * inv;
    }
}

extern "C" void kernel_launch(void* const* d_in, const int* in_sizes, int n_in,
                              void* d_out, int out_size, void* d_ws, size_t ws_size,
                              hipStream_t stream) {
    (void)in_sizes; (void)n_in; (void)out_size; (void)d_ws; (void)ws_size;
    const float* Q    = (const float*)d_in[0];
    const float* K    = (const float*)d_in[1];
    const float* V    = (const float*)d_in[2];
    const int*   mask = (const int*)d_in[3];
    float*       out  = (float*)d_out;

    const int blocks = (BATCH * (SEQ / 16)) / WAVES;   // 512
    fa_causal_wmma_kernel<<<blocks, WAVES * 32, 0, stream>>>(Q, K, V, mask, out);
}
